// BaseHSMM_9749575762404
// MI455X (gfx1250) — matrix operand order repaired and checked by reading, not compiled, not run on hardware
//
#include <hip/hip_runtime.h>
#include <hip/hip_bf16.h>

typedef __attribute__((ext_vector_type(16))) _Float16 v16h;
typedef __attribute__((ext_vector_type(8)))  float    v8f;

#define HSMM_B    16
#define HSMM_T    2048
#define HSMM_K    256
#define HSMM_DMAX 64
#define NEGINF   (-1e30f)
#define L2E       1.4426950408889634f   // log2(e)
#define LN2       0.6931471805599453f   // ln(2)

// One workgroup per batch, 256 threads (8 waves, wave32). Thread j owns state j.
// Entire recurrence runs in base-2 log domain (native v_exp_f32/v_log_f32 units).
__global__ __launch_bounds__(256, 1)
void hsmm_forward_kernel(const float* __restrict__ logB,     // [B,T,K]
                         const float* __restrict__ pi,       // [K]
                         const float* __restrict__ Alg,      // [K,K]
                         const float* __restrict__ Dlg,      // [K,Dmax]
                         float* __restrict__ loglik,         // [B]
                         float* __restrict__ alphas)         // [B,T,K]
{
    __shared__ _Float16 PT[HSMM_K * HSMM_K];      // P^T: PT[j*K+i] = exp(A[i][j]) (128 KB)
    __shared__ _Float16 ea16[2][HSMM_K + 16];     // exp2(alpha2 - m_stale), parity-buffered
    __shared__ float    lds_ain[2][HSMM_K];       // WMMA dot products, parity-buffered
    __shared__ float    lds_red[2][8];            // per-wave alpha2 max, parity-buffered

    const int tid  = threadIdx.x;          // 0..255  == state j
    const int j    = tid;
    const int b    = blockIdx.x;           // batch
    const int wv   = tid >> 5;             // wave id 0..7
    const int lane = tid & 31;
    const int nlo  = lane & 15;
    const int hiH  = (lane >> 4) & 1;      // half-wave selector for WMMA operand layout

    const size_t bbase = (size_t)b * HSMM_T * HSMM_K;

    // ---- Preamble: P^T = exp(A)^T into LDS (f16) ----
    for (int idx = tid; idx < HSMM_K * HSMM_K; idx += 256) {
        int i  = idx >> 8;        // row of A
        int jj = idx & 255;       // col of A
        PT[jj * HSMM_K + i] = (_Float16)__builtin_amdgcn_exp2f(Alg[idx] * L2E);
    }

    // ---- Per-thread state (base-2 scaled), all in VGPRs ----
    float dreg[HSMM_DMAX];
    float fbuf[HSMM_DMAX];     // fbuf[d] = log2e*(a_{t-(d+1)}(j) - cumb[t-(d+1)](j))
#pragma unroll
    for (int d = 0; d < HSMM_DMAX; ++d) {
        dreg[d] = Dlg[j * HSMM_DMAX + d] * L2E;
        fbuf[d] = NEGINF;
    }
    fbuf[0] = pi[j] * L2E;     // t=1: only d=1 valid, a_0 = pi, cumb_0 = 0

    float cum2    = 0.0f;                  // log2e * cumb_t(j)
    float lb_next = logB[bbase + j];       // software-pipelined emission load
    float alpha2  = NEGINF;
    float mstale  = 0.0f;                  // one-step-stale block max of alpha2

    __syncthreads();

    for (int t = 0; t < HSMM_T; ++t) {
        const int par = t & 1;

        // pipeline next emission row; prefetch ahead (global_prefetch_b8)
        float lb = lb_next;
        int tn = (t + 1  < HSMM_T) ? (t + 1)  : (HSMM_T - 1);
        int tp = (t + 16 < HSMM_T) ? (t + 16) : (HSMM_T - 1);
        lb_next = logB[bbase + (size_t)tn * HSMM_K + j];
        __builtin_prefetch(&logB[bbase + (size_t)tp * HSMM_K + j], 0, 1);
        cum2 = fmaf(lb, L2E, cum2);

        // ---- (1) duration logsumexp, register-only ----
        float mx = -3.0e38f;
#pragma unroll
        for (int d = 0; d < HSMM_DMAX; ++d)
            mx = fmaxf(mx, fbuf[d] + dreg[d]);
        float s = 0.0f;
#pragma unroll
        for (int d = 0; d < HSMM_DMAX; ++d)
            s += __builtin_amdgcn_exp2f(fbuf[d] + dreg[d] - mx);
        alpha2 = mx + __builtin_amdgcn_logf(s) + cum2;

        // ---- emit alpha_t(j) (convert back to natural log) ----
        alphas[bbase + (size_t)t * HSMM_K + j] = alpha2 * LN2;

        // ---- (2) publish wave max (consumed NEXT step) + staged exp row ----
        float wm = alpha2;
        wm = fmaxf(wm, __shfl_xor(wm, 1, 32));
        wm = fmaxf(wm, __shfl_xor(wm, 2, 32));
        wm = fmaxf(wm, __shfl_xor(wm, 4, 32));
        wm = fmaxf(wm, __shfl_xor(wm, 8, 32));
        wm = fmaxf(wm, __shfl_xor(wm, 16, 32));
        if (lane == 0) lds_red[par][wv] = wm;
        if (t > 0)     // stale shift: bounded per-step drift, ample f16 headroom
            ea16[par][j] = (_Float16)__builtin_amdgcn_exp2f(alpha2 - mstale);
        __syncthreads();                               // barrier A

        if (t == 0) {  // exact shift for the first step only
            float m0 = lds_red[0][0];
#pragma unroll
            for (int w = 1; w < 8; ++w) m0 = fmaxf(m0, lds_red[0][w]);
            mstale = m0;
            ea16[0][j] = (_Float16)__builtin_amdgcn_exp2f(alpha2 - mstale);
            __syncthreads();
        }

        // ---- (3) transition GEMV in exp domain via WMMA f32_16x16x32_f16 ----
        const _Float16* ea = ea16[par];
        v16h av[8];
#pragma unroll
        for (int kk = 0; kk < 8; ++kk) {
            const int ab = kk * 32 + hiH * 16;
#pragma unroll
            for (int h = 0; h < 8; ++h) {
                av[kk][h]     = ea[ab + h];
                av[kk][h + 8] = ea[ab + 8 + h];
            }
        }
#pragma unroll
        for (int tt = 0; tt < 2; ++tt) {               // tiles wv and wv+8 (static trip)
            const int j0 = (wv + tt * 8) << 4;
            v8f acc = {};
#pragma unroll
            for (int kk = 0; kk < 8; ++kk) {
                v16h bv;
                const _Float16* prow = &PT[(size_t)(j0 + nlo) * HSMM_K + kk * 32 + hiH * 16];
#pragma unroll
                for (int h = 0; h < 16; ++h) bv[h] = prow[h];
                acc = __builtin_amdgcn_wmma_f32_16x16x32_f16(
                        false, av[kk], false, bv, (short)0, acc, false, false);
            }
            lds_ain[par][j0 + nlo] = acc[0];           // rows identical (broadcast A)
        }
        __syncthreads();                               // barrier B

        // ---- (4) back to log domain, advance stale max, shift duration buffer ----
        float dot   = fmaxf(lds_ain[par][j], 1e-38f);  // branchless zero guard
        float a_in2 = mstale + __builtin_amdgcn_logf(dot);
        float mnew  = lds_red[par][0];
#pragma unroll
        for (int w = 1; w < 8; ++w) mnew = fmaxf(mnew, lds_red[par][w]);
        mstale = mnew;
#pragma unroll
        for (int d = HSMM_DMAX - 1; d > 0; --d) fbuf[d] = fbuf[d - 1];
        fbuf[0] = a_in2 - cum2;
    }

    // ---- loglik[b] = ln-logsumexp_j alpha_{T-1}(j) ----
    float wm = alpha2;
    wm = fmaxf(wm, __shfl_xor(wm, 1, 32));
    wm = fmaxf(wm, __shfl_xor(wm, 2, 32));
    wm = fmaxf(wm, __shfl_xor(wm, 4, 32));
    wm = fmaxf(wm, __shfl_xor(wm, 8, 32));
    wm = fmaxf(wm, __shfl_xor(wm, 16, 32));
    if (lane == 0) lds_red[0][wv] = wm;
    __syncthreads();
    float m2 = lds_red[0][0];
#pragma unroll
    for (int w = 1; w < 8; ++w) m2 = fmaxf(m2, lds_red[0][w]);
    __syncthreads();

    float e = __builtin_amdgcn_exp2f(alpha2 - m2);
    e += __shfl_xor(e, 1, 32);
    e += __shfl_xor(e, 2, 32);
    e += __shfl_xor(e, 4, 32);
    e += __shfl_xor(e, 8, 32);
    e += __shfl_xor(e, 16, 32);
    if (lane == 0) lds_red[0][wv] = e;
    __syncthreads();
    if (tid == 0) {
        float stot = 0.0f;
#pragma unroll
        for (int w = 0; w < 8; ++w) stot += lds_red[0][w];
        loglik[b] = (m2 + __builtin_amdgcn_logf(stot)) * LN2;
    }
}

extern "C" void kernel_launch(void* const* d_in, const int* in_sizes, int n_in,
                              void* d_out, int out_size, void* d_ws, size_t ws_size,
                              hipStream_t stream) {
    const float* logB = (const float*)d_in[0];   // [16,2048,256]
    const float* pi   = (const float*)d_in[1];   // [256]
    const float* Alg  = (const float*)d_in[2];   // [256,256]
    const float* Dlg  = (const float*)d_in[3];   // [256,64]

    float* out    = (float*)d_out;
    float* loglik = out;                 // [16]
    float* alphas = out + HSMM_B;        // [16,2048,256]

    hsmm_forward_kernel<<<dim3(HSMM_B), dim3(HSMM_K), 0, stream>>>(
        logB, pi, Alg, Dlg, loglik, alphas);
}
